// CGCNNModel_41240275976350
// MI455X (gfx1250) — compile-verified, matrix operand order checked by reading, not run on hardware
//
#include <hip/hip_runtime.h>
#include <math.h>

// ---------------- problem constants (match reference) ----------------
static constexpr int S    = 128;
static constexpr int N    = 512;
static constexpr int P    = 92;
static constexpr int DIN  = P + N;   // 604
static constexpr int F    = 128;     // DIMS[0..2]
static constexpr int HID  = 256;
static constexpr int H0   = 100;

// ---------------- WMMA types ----------------
typedef __attribute__((ext_vector_type(16))) __bf16 v16bf;
typedef __attribute__((ext_vector_type(8)))  float  v8f;

union FragBF {
    v16bf v;
    uint4 q[2];
};
union Run8 {
    unsigned short u[8];
    uint4 q;
};

__device__ __forceinline__ unsigned short f2bf(float f) {
    unsigned int u = __float_as_uint(f);
    u += 0x7fffu + ((u >> 16) & 1u);          // round-to-nearest-even
    return (unsigned short)(u >> 16);
}
__device__ __forceinline__ float bf2f(unsigned short v) {
    return __uint_as_float(((unsigned int)v) << 16);
}

// ---------------- degree / d^{-1/2} ----------------
__global__ void deg_kernel(const int* __restrict__ conn, float* __restrict__ dinv) {
    int wid  = threadIdx.x >> 5;
    int lane = threadIdx.x & 31;
    int row  = blockIdx.x * 8 + wid;          // row in [0, S*N)
    const int* r = conn + (size_t)row * N;
    int cnt = 0;
    for (int j = lane; j < N; j += 32) cnt += (r[j] != 0) ? 1 : 0;
    #pragma unroll
    for (int off = 16; off > 0; off >>= 1) cnt += __shfl_xor(cnt, off, 32);
    if (lane == 0) dinv[row] = rsqrtf((float)(cnt + 1));
}

// ---------------- bf16 WMMA GEMM, 128x128 block tile, BK=32 ----------------
// LDS tiles live in WMMA fragment order: slot = (tile*32 + lane)*16 + i, so a
// lane's operand is 32 contiguous aligned bytes -> 2x ds_load_b128.
// A-operand (16x32 bf16): lane -> M = lane&15, half = lane>>4;
//   element i -> k = (i<8) ? half*8+i : 16 + half*8 + (i-8)
// staging inverse: half=(k>>3)&1, i=(k&7)+8*(k>>4), lane=row+16*half.
//
// MODE 0: A = concat(node_feat,bond_feat) f32, B = W f32, out = Yt bf16 (transposed, [f][node])
// MODE 1: A = Xb bf16 row-major,            B = W f32, out = Yt bf16 (transposed)
// MODE 2: A = An from conn+dinv,            B = Yt bf16 (transposed), out = Xb row-major (+bias)
template <int MODE>
__device__ __forceinline__ void load_tile(int kt, int kh, int gm, int bn, int s,
                                          const float* __restrict__ nf,
                                          const float* __restrict__ bfeat,
                                          const int*   __restrict__ conn,
                                          const float* __restrict__ dinv,
                                          const unsigned short* __restrict__ Abf,
                                          const float* __restrict__ Wf,
                                          const unsigned short* __restrict__ Bbf,
                                          int Kreal,
                                          Run8& a0, Run8& a1, Run8& b0, Run8& b1) {
    const int lo = kt + kh;                   // wave-uniform k base of this 16-run pair? (kh uniform per wave)
    // ---- A ----
    if (MODE == 0) {
        if (lo + 16 <= P) {                   // fully node_feat
            const float* ap = &nf[((size_t)s * N + gm) * P + lo];
            #pragma unroll
            for (int j = 0; j < 8; ++j) { a0.u[j] = f2bf(ap[j]); a1.u[j] = f2bf(ap[8 + j]); }
        } else if (lo >= P && lo + 16 <= DIN) {   // fully bond_feat
            const float* ap = &bfeat[((size_t)s * N + gm) * N + (lo - P)];
            #pragma unroll
            for (int j = 0; j < 8; ++j) { a0.u[j] = f2bf(ap[j]); a1.u[j] = f2bf(ap[8 + j]); }
        } else {                              // straddle / tail tiles (2 of 38)
            #pragma unroll
            for (int j = 0; j < 8; ++j) {
                int k0 = lo + j, k1 = k0 + 8;
                float v0 = 0.f, v1 = 0.f;
                if (k0 < P)        v0 = nf[((size_t)s * N + gm) * P + k0];
                else if (k0 < DIN) v0 = bfeat[((size_t)s * N + gm) * N + (k0 - P)];
                if (k1 < P)        v1 = nf[((size_t)s * N + gm) * P + k1];
                else if (k1 < DIN) v1 = bfeat[((size_t)s * N + gm) * N + (k1 - P)];
                a0.u[j] = f2bf(v0); a1.u[j] = f2bf(v1);
            }
        }
    } else if (MODE == 1) {
        const uint4* ap = (const uint4*)&Abf[((size_t)s * N + gm) * F + lo];
        a0.q = ap[0];
        a1.q = ap[1];
    } else {
        const float di = dinv[s * N + gm];
        const int* cp = &conn[((size_t)s * N + gm) * N + lo];
        const float* dp = &dinv[s * N + lo];
        #pragma unroll
        for (int j = 0; j < 8; ++j) {
            int k0 = lo + j, k1 = k0 + 8;
            float v0 = ((cp[j]     != 0) ? 1.f : 0.f) + ((gm == k0) ? 1.f : 0.f);
            float v1 = ((cp[8 + j] != 0) ? 1.f : 0.f) + ((gm == k1) ? 1.f : 0.f);
            a0.u[j] = f2bf(v0 * di * dp[j]);
            a1.u[j] = f2bf(v1 * di * dp[8 + j]);
        }
    }
    // ---- B ----
    if (MODE == 2) {                          // Yt transposed: [f][node] contiguous in k
        const uint4* bp = (const uint4*)&Bbf[((size_t)s * F + bn) * N + lo];
        b0.q = bp[0];
        b1.q = bp[1];
    } else {
        #pragma unroll
        for (int j = 0; j < 8; ++j) {
            int k0 = lo + j, k1 = k0 + 8;
            b0.u[j] = f2bf((k0 < Kreal) ? Wf[(size_t)k0 * F + bn] : 0.f);
            b1.u[j] = f2bf((k1 < Kreal) ? Wf[(size_t)k1 * F + bn] : 0.f);
        }
    }
}

template <int MODE>
__global__ void gemm_kernel(const float* __restrict__ nf,
                            const float* __restrict__ bfeat,
                            const int*   __restrict__ conn,
                            const float* __restrict__ dinv,
                            const unsigned short* __restrict__ Abf,
                            const float* __restrict__ Wf,
                            const unsigned short* __restrict__ Bbf,
                            const float* __restrict__ bias,
                            unsigned short* __restrict__ outb,
                            int Kpad, int Kreal) {
    __shared__ unsigned short AF[2 * 8 * 32 * 16];   // double-buffered, frag order, 2x8 KB
    __shared__ unsigned short BF[2 * 8 * 32 * 16];

    const int s     = blockIdx.x;
    const int mBase = blockIdx.y * 128;
    const int tid   = threadIdx.x;
    const int lane  = tid & 31;
    const int wid   = tid >> 5;               // 8 waves, wave w -> rows [16w,16w+16)

    v8f acc[8];
    const v8f vzero = {0.f, 0.f, 0.f, 0.f, 0.f, 0.f, 0.f, 0.f};
    #pragma unroll
    for (int t = 0; t < 8; ++t) acc[t] = vzero;

    // staging: thread owns A-row am and B-col bn; k-half is wave-uniform so
    // MODE 0 region selection does not diverge within a wave.
    const int am    = tid & 127;              // A tile row
    const int bn    = tid & 127;              // B tile col
    const int kh    = (tid >> 7) * 16;        // k half: 0 or 16 (uniform per wave)
    const int ibase = kh ? 8 : 0;
    const int awt   = am >> 4, aml = am & 15;
    const int bct   = bn >> 4, bnl = bn & 15;
    const int gm    = mBase + am;

    // LDS destinations (ushort indices) for the two 8-element runs
    const int aslot0 = (awt * 32 + aml)      * 16 + ibase;
    const int aslot1 = (awt * 32 + aml + 16) * 16 + ibase;
    const int bslot0 = (bct * 32 + bnl)      * 16 + ibase;
    const int bslot1 = (bct * 32 + bnl + 16) * 16 + ibase;

    // prologue: stage tile 0 into buffer 0
    {
        Run8 a0, a1, b0, b1;
        load_tile<MODE>(0, kh, gm, bn, s, nf, bfeat, conn, dinv, Abf, Wf, Bbf, Kreal,
                        a0, a1, b0, b1);
        *(uint4*)&AF[aslot0] = a0.q;  *(uint4*)&AF[aslot1] = a1.q;
        *(uint4*)&BF[bslot0] = b0.q;  *(uint4*)&BF[bslot1] = b1.q;
    }
    __syncthreads();

    int cur = 0;
    for (int kt = 0; kt < Kpad; kt += 32) {
        const bool has_next = (kt + 32) < Kpad;   // uniform
        Run8 a0, a1, b0, b1;
        if (has_next) {
            load_tile<MODE>(kt + 32, kh, gm, bn, s, nf, bfeat, conn, dinv, Abf, Wf, Bbf,
                            Kreal, a0, a1, b0, b1);
        }

        // ---- compute from buffer `cur` ----
        const int co = cur * 4096;
        FragBF a;
        {
            const uint4* ap = (const uint4*)&AF[co + (wid * 32 + lane) * 16];
            a.q[0] = ap[0];
            a.q[1] = ap[1];
        }
        #pragma unroll
        for (int t = 0; t < 8; ++t) {
            FragBF b;
            const uint4* bp = (const uint4*)&BF[co + (t * 32 + lane) * 16];
            b.q[0] = bp[0];
            b.q[1] = bp[1];
            acc[t] = __builtin_amdgcn_wmma_f32_16x16x32_bf16(
                false, a.v, false, b.v, (short)0, acc[t], false, false);
        }

        if (has_next) {
            const int no = (cur ^ 1) * 4096;
            *(uint4*)&AF[no + aslot0] = a0.q;  *(uint4*)&AF[no + aslot1] = a1.q;
            *(uint4*)&BF[no + bslot0] = b0.q;  *(uint4*)&BF[no + bslot1] = b1.q;
        }
        __syncthreads();
        cur ^= 1;
    }

    // ---- store D: VGPR r -> M = half*8 + r, N = lane&15 ----
    const int lrow = lane & 15;
    const int half = lane >> 4;
    if (MODE == 2) {
        // row-major Xb (+bias): scalar bf16 stores
        #pragma unroll
        for (int t = 0; t < 8; ++t) {
            #pragma unroll
            for (int r = 0; r < 8; ++r) {
                int row = mBase + wid * 16 + half * 8 + r;
                int col = t * 16 + lrow;
                outb[((size_t)s * N + row) * F + col] = f2bf(acc[t][r] + bias[col]);
            }
        }
    } else {
        // transposed Yt [f][node]: 8 consecutive rows per accumulator -> uint4 store
        #pragma unroll
        for (int t = 0; t < 8; ++t) {
            int col = t * 16 + lrow;                      // feature f
            int row0 = mBase + wid * 16 + half * 8;       // first of 8 consecutive nodes
            Run8 r8;
            #pragma unroll
            for (int r = 0; r < 8; ++r) r8.u[r] = f2bf(acc[t][r]);
            *(uint4*)&outb[((size_t)s * F + col) * N + row0] = r8.q;
        }
    }
}

// ---------------- pool (segment sum) + softplus + MLP heads ----------------
__global__ void pool_head_kernel(const unsigned short* __restrict__ X,  // (S,N,F) bf16
                                 const float* __restrict__ Wh, const float* __restrict__ bh,
                                 const float* __restrict__ Wi, const float* __restrict__ bi,
                                 const float* __restrict__ We, const float* __restrict__ be,
                                 const float* __restrict__ Wlb, const float* __restrict__ blb,
                                 const float* __restrict__ Wub, const float* __restrict__ bub,
                                 float* __restrict__ out) {
    __shared__ float sp[F];
    __shared__ float h[HID];
    const int s = blockIdx.x;
    const int t = threadIdx.x;          // 256 threads

    if (t < F) {
        float acc = 0.f;
        const unsigned short* p = X + (size_t)s * N * F + t;
        for (int n = 0; n < N; ++n) acc += bf2f(p[(size_t)n * F]);
        sp[t] = (acc > 20.f) ? acc : log1pf(expf(acc));   // softplus
    }
    __syncthreads();

    {   // h = softplus(pooled) @ Wh + bh   (128 -> 256)
        float acc = bh[t];
        #pragma unroll 4
        for (int f = 0; f < F; ++f) acc += sp[f] * Wh[f * HID + t];
        h[t] = acc;
    }
    __syncthreads();

    if (t < H0) {
        float a = bi[t];
        for (int j = 0; j < HID; ++j) a += h[j] * Wi[j * H0 + t];
        out[s * H0 + t] = a;
    } else if (t == H0) {
        float a = be[0];
        for (int j = 0; j < HID; ++j) a += h[j] * We[j];
        out[S * H0 + s] = a;
    } else if (t == H0 + 1) {
        float a = blb[0];
        for (int j = 0; j < HID; ++j) a += h[j] * Wlb[j];
        out[S * H0 + S + s] = a;
    } else if (t == H0 + 2) {
        float a = bub[0];
        for (int j = 0; j < HID; ++j) a += h[j] * Wub[j];
        out[S * H0 + 2 * S + s] = a;
    }
}

// ---------------- host-side orchestration ----------------
extern "C" void kernel_launch(void* const* d_in, const int* in_sizes, int n_in,
                              void* d_out, int out_size, void* d_ws, size_t ws_size,
                              hipStream_t stream) {
    (void)in_sizes; (void)n_in; (void)out_size; (void)ws_size;

    const float* nf    = (const float*)d_in[0];   // (S,N,P)
    const float* bfeat = (const float*)d_in[1];   // (S,N,N)
    const int*   conn  = (const int*)  d_in[2];   // (S,N,N)
    // d_in[3] = batchAssign (implicit in layout)
    const float* W0  = (const float*)d_in[4];  const float* b0  = (const float*)d_in[5];
    const float* W1  = (const float*)d_in[6];  const float* b1  = (const float*)d_in[7];
    const float* W2  = (const float*)d_in[8];  const float* b2  = (const float*)d_in[9];
    const float* Wh  = (const float*)d_in[10]; const float* bh  = (const float*)d_in[11];
    const float* Wi  = (const float*)d_in[12]; const float* bi  = (const float*)d_in[13];
    const float* We  = (const float*)d_in[14]; const float* be  = (const float*)d_in[15];
    const float* Wlb = (const float*)d_in[16]; const float* blb = (const float*)d_in[17];
    const float* Wub = (const float*)d_in[18]; const float* bub = (const float*)d_in[19];
    float* out = (float*)d_out;

    // workspace layout
    char* ws = (char*)d_ws;
    float*          dinv = (float*)ws;                                   //  S*N*4   = 256 KB
    unsigned short* Xb   = (unsigned short*)(ws + (size_t)S * N * 4);    //  row-major (S,N,F), 16 MB
    unsigned short* Yt   = Xb + (size_t)S * N * F;                       //  transposed (S,F,N), 16 MB

    const dim3 gGemm(S, N / 128);

    // d^{-1/2}
    deg_kernel<<<(S * N) / 8, 256, 0, stream>>>(conn, dinv);

    // layer 0: Yt = (concat @ W0)^T ; Xb = An @ Y + b0
    gemm_kernel<0><<<gGemm, 256, 0, stream>>>(nf, bfeat, nullptr, nullptr, nullptr,
                                              W0, nullptr, nullptr, Yt, 608, DIN);
    gemm_kernel<2><<<gGemm, 256, 0, stream>>>(nullptr, nullptr, conn, dinv, nullptr,
                                              nullptr, Yt, b0, Xb, N, N);
    // layer 1
    gemm_kernel<1><<<gGemm, 256, 0, stream>>>(nullptr, nullptr, nullptr, nullptr, Xb,
                                              W1, nullptr, nullptr, Yt, F, F);
    gemm_kernel<2><<<gGemm, 256, 0, stream>>>(nullptr, nullptr, conn, dinv, nullptr,
                                              nullptr, Yt, b1, Xb, N, N);
    // layer 2
    gemm_kernel<1><<<gGemm, 256, 0, stream>>>(nullptr, nullptr, nullptr, nullptr, Xb,
                                              W2, nullptr, nullptr, Yt, F, F);
    gemm_kernel<2><<<gGemm, 256, 0, stream>>>(nullptr, nullptr, conn, dinv, nullptr,
                                              nullptr, Yt, b2, Xb, N, N);

    // pool + softplus + heads
    pool_head_kernel<<<S, 256, 0, stream>>>(Xb, Wh, bh, Wi, bi, We, be,
                                            Wlb, blb, Wub, bub, out);
}
